// LocationAwareAttention_64364379898136
// MI455X (gfx1250) — compile-verified, matrix operand order checked by reading, not run on hardware
//
#include <hip/hip_runtime.h>
#include <hip/hip_bf16.h>

// ---------------------------------------------------------------------------
// Location-aware attention (Tacotron2 style) for MI455X / gfx1250.
// B=32, T=4096, H=512, A=512, C=32
// Heavy einsum (bth,ah->bta fused with tanh+dot+reduce) runs on
// v_wmma_f32_16x16x32_bf16.
// ---------------------------------------------------------------------------

#define B_  32
#define T_  4096
#define H_  512
#define A_  512
#define C_  32

typedef __bf16 bf16_t;
typedef __attribute__((ext_vector_type(16))) __bf16 v16bf;
typedef __attribute__((ext_vector_type(8)))  __bf16 v8bf;
typedef __attribute__((ext_vector_type(8)))  float  v8f;

// ---- fragment loaders -----------------------------------------------------
// 16-bit A/B fragment, 16x32 (MxK) layout per CDNA5 ISA:
//   lanes 0-15 : row = lane,    K = {0..7, 16..23}
//   lanes16-31 : row = lane-16, K = {8..15, 24..31}
// Caller passes p = rowbase + (lane&15)*stride + k0 + ((lane<16)?0:8).

__device__ __forceinline__ v16bf load_frag_bf(const bf16_t* __restrict__ p) {
    v8bf lo = *reinterpret_cast<const v8bf*>(p);
    v8bf hi = *reinterpret_cast<const v8bf*>(p + 16);
    v16bf f;
#pragma unroll
    for (int i = 0; i < 8; ++i) { f[i] = lo[i]; f[i + 8] = hi[i]; }
    return f;
}

__device__ __forceinline__ v16bf load_frag_f32(const float* __restrict__ p) {
    v16bf f;
#pragma unroll
    for (int i = 0; i < 8; ++i) {
        f[i]     = (bf16_t)p[i];
        f[i + 8] = (bf16_t)p[i + 16];
    }
    return f;
}

// ---- prep kernels ---------------------------------------------------------

__global__ void dec_proj_kernel(const float* __restrict__ dec,   // [B,1,H]
                                const float* __restrict__ W,     // [A,H]
                                const float* __restrict__ bias,  // [A]
                                float* __restrict__ decb)        // [B,A]
{
    int idx = blockIdx.x * blockDim.x + threadIdx.x;
    if (idx >= B_ * A_) return;
    int b = idx >> 9, a = idx & (A_ - 1);
    const float* wr = W + (size_t)a * H_;
    const float* dr = dec + (size_t)b * H_;
    float s = bias[a];
#pragma unroll 8
    for (int h = 0; h < H_; ++h) s += wr[h] * dr[h];
    decb[idx] = s;
}

__global__ void cvt_bf16_kernel(const float* __restrict__ in,
                                bf16_t* __restrict__ out, int n)
{
    int i = blockIdx.x * blockDim.x + threadIdx.x;
    if (i < n) out[i] = (bf16_t)in[i];
}

__global__ void conv_feat_kernel(const float* __restrict__ align,  // [B,T]
                                 const float* __restrict__ cw,     // [C,1,3]
                                 const float* __restrict__ cb,     // [C]
                                 bf16_t* __restrict__ convbf)      // [B,T,C]
{
    int idx = blockIdx.x * blockDim.x + threadIdx.x;
    if (idx >= B_ * T_ * C_) return;
    int c = idx & (C_ - 1);
    int t = (idx >> 5) & (T_ - 1);
    int b = idx >> 17;
    const float* al = align + (size_t)b * T_;
    float s = cb[c];
    float w0 = cw[c * 3 + 0], w1 = cw[c * 3 + 1], w2 = cw[c * 3 + 2];
    if (t > 0)      s += w0 * al[t - 1];
    s += w1 * al[t];
    if (t < T_ - 1) s += w2 * al[t + 1];
    convbf[idx] = (bf16_t)s;
}

// ---- main fused scores kernel --------------------------------------------
// grid (T/64, B), 256 threads (8 waves).
// Block: 64 t-rows (4 M-tiles). Wave w: A-columns [w*64, w*64+64) (4 N-tiles).
// scores[b,t] = sum_a w[a]*tanh( enc_proj + loc_proj + decb )

__global__ __launch_bounds__(256)
void attn_scores_kernel(const float*  __restrict__ enc,     // [B,T,H]
                        const bf16_t* __restrict__ Vbf,     // [A,H]
                        const bf16_t* __restrict__ Ubf,     // [A,C]
                        const bf16_t* __restrict__ convbf,  // [B,T,C]
                        const float*  __restrict__ decb,    // [B,A]
                        const float*  __restrict__ wvec,    // [A]
                        float* __restrict__ scores)         // [B,T]
{
    __shared__ float sc[64];

    const int lane = threadIdx.x & 31;
    const int wave = threadIdx.x >> 5;
    const int b    = blockIdx.y;
    const int t0   = blockIdx.x * 64;
    const int n0   = wave * 64;
    const int lrow = lane & 15;
    const int kb   = (lane < 16) ? 0 : 8;   // K-half select
    const int hi   = lane >> 4;             // accumulator row-half

    const float* encB = enc + ((size_t)b * T_) * H_;
    const bf16_t* cvB = convbf + ((size_t)b * T_) * C_;

    v8f acc[4][4];
#pragma unroll
    for (int mt = 0; mt < 4; ++mt)
#pragma unroll
        for (int nt = 0; nt < 4; ++nt)
#pragma unroll
            for (int r = 0; r < 8; ++r) acc[mt][nt][r] = 0.0f;

    // ---- K loop over H (enc_proj: V[a,h] * enc[b,t,h]) ----
    for (int k = 0; k < H_ / 32; ++k) {
        const int h0 = k * 32 + kb;
        v16bf bf[4];
#pragma unroll
        for (int nt = 0; nt < 4; ++nt)
            bf[nt] = load_frag_bf(Vbf + (size_t)(n0 + nt * 16 + lrow) * H_ + h0);
#pragma unroll
        for (int mt = 0; mt < 4; ++mt) {
            v16bf af = load_frag_f32(encB + (size_t)(t0 + mt * 16 + lrow) * H_ + h0);
#pragma unroll
            for (int nt = 0; nt < 4; ++nt)
                acc[mt][nt] = __builtin_amdgcn_wmma_f32_16x16x32_bf16(
                    false, af, false, bf[nt], (short)0, acc[mt][nt], false, false);
        }
    }

    // ---- one extra K=32 step for loc_proj: U[a,c] * conv_feat[b,t,c] ----
    {
        v16bf bf[4];
#pragma unroll
        for (int nt = 0; nt < 4; ++nt)
            bf[nt] = load_frag_bf(Ubf + (size_t)(n0 + nt * 16 + lrow) * C_ + kb);
#pragma unroll
        for (int mt = 0; mt < 4; ++mt) {
            v16bf af = load_frag_bf(cvB + (size_t)(t0 + mt * 16 + lrow) * C_ + kb);
#pragma unroll
            for (int nt = 0; nt < 4; ++nt)
                acc[mt][nt] = __builtin_amdgcn_wmma_f32_16x16x32_bf16(
                    false, af, false, bf[nt], (short)0, acc[mt][nt], false, false);
        }
    }

    // ---- epilogue: + decb, tanh, *w, reduce over A ----
    if (threadIdx.x < 64) sc[threadIdx.x] = 0.0f;
    __syncthreads();

#pragma unroll
    for (int mt = 0; mt < 4; ++mt) {
#pragma unroll
        for (int nt = 0; nt < 4; ++nt) {
            const int aidx = n0 + nt * 16 + lrow;        // N column
            const float dv = decb[b * A_ + aidx];
            const float wv = wvec[aidx];
#pragma unroll
            for (int r = 0; r < 8; ++r) {
                // C layout: VGPR r, lanes 0-15 -> M=r ; lanes 16-31 -> M=r+8
                float s = wv * tanhf(acc[mt][nt][r] + dv);
                // butterfly reduce over the 16-lane half (N direction)
                s += __shfl_xor(s, 1, 32);
                s += __shfl_xor(s, 2, 32);
                s += __shfl_xor(s, 4, 32);
                s += __shfl_xor(s, 8, 32);
                if (lrow == 0)
                    atomicAdd(&sc[mt * 16 + r + 8 * hi], s);
            }
        }
    }
    __syncthreads();
    if (threadIdx.x < 64)
        scores[(size_t)b * T_ + t0 + threadIdx.x] = sc[threadIdx.x];
}

// ---- softmax over T per batch --------------------------------------------

__global__ __launch_bounds__(256)
void softmax_kernel(const float* __restrict__ scores,  // [B,T]
                    float* __restrict__ align_out)     // [B,T]
{
    __shared__ float red[256];
    const int b = blockIdx.x;
    const int tid = threadIdx.x;
    const float* s = scores + (size_t)b * T_;

    float m = -3.0e38f;
    for (int t = tid; t < T_; t += 256) m = fmaxf(m, s[t]);
    red[tid] = m; __syncthreads();
    for (int off = 128; off > 0; off >>= 1) {
        if (tid < off) red[tid] = fmaxf(red[tid], red[tid + off]);
        __syncthreads();
    }
    m = red[0]; __syncthreads();

    float sum = 0.0f;
    for (int t = tid; t < T_; t += 256) sum += __expf(s[t] - m);
    red[tid] = sum; __syncthreads();
    for (int off = 128; off > 0; off >>= 1) {
        if (tid < off) red[tid] += red[tid + off];
        __syncthreads();
    }
    sum = red[0]; __syncthreads();

    const float inv = 1.0f / sum;
    for (int t = tid; t < T_; t += 256)
        align_out[(size_t)b * T_ + t] = __expf(s[t] - m) * inv;
}

// ---- weighted sum: out[b,h] = sum_t align[b,t] * enc[b,t,h] ---------------

__global__ void zero_out_kernel(float* __restrict__ out)
{
    int i = blockIdx.x * blockDim.x + threadIdx.x;
    if (i < B_ * H_) out[i] = 0.0f;
}

__global__ __launch_bounds__(512)
void weighted_sum_kernel(const float* __restrict__ enc,    // [B,T,H]
                         const float* __restrict__ align,  // [B,T]
                         float* __restrict__ out)          // [B,H]
{
    const int b = blockIdx.x;
    const int chunk = blockIdx.y;          // 8 chunks of 512 t-rows
    const int h = threadIdx.x;
    const float* e = enc + ((size_t)b * T_ + (size_t)chunk * 512) * H_;
    const float* a = align + (size_t)b * T_ + (size_t)chunk * 512;
    float acc = 0.0f;
    for (int t = 0; t < 512; ++t)
        acc += a[t] * e[(size_t)t * H_ + h];
    atomicAdd(&out[b * H_ + h], acc);
}

// ---------------------------------------------------------------------------

extern "C" void kernel_launch(void* const* d_in, const int* in_sizes, int n_in,
                              void* d_out, int out_size, void* d_ws, size_t ws_size,
                              hipStream_t stream)
{
    const float* dec    = (const float*)d_in[0];  // [B,1,H]
    const float* enc    = (const float*)d_in[1];  // [B,T,H]
    const float* lalign = (const float*)d_in[2];  // [B,T]
    const float* conv_w = (const float*)d_in[3];  // [C,1,3]
    const float* conv_b = (const float*)d_in[4];  // [C]
    const float* W      = (const float*)d_in[5];  // [A,H]
    const float* V      = (const float*)d_in[6];  // [A,H]
    const float* U      = (const float*)d_in[7];  // [A,C]
    const float* bvec   = (const float*)d_in[8];  // [A]
    const float* wvec   = (const float*)d_in[9];  // [A]

    float* out_vec   = (float*)d_out;                    // [B,H]
    float* align_out = (float*)d_out + (size_t)B_ * H_;  // [B,T]

    // workspace layout
    char* ws = (char*)d_ws;
    size_t off = 0;
    float*  decb   = (float*)(ws + off);  off += (size_t)B_ * A_ * sizeof(float);     // 64 KB
    bf16_t* Vbf    = (bf16_t*)(ws + off); off += (size_t)A_ * H_ * sizeof(bf16_t);    // 512 KB
    bf16_t* Ubf    = (bf16_t*)(ws + off); off += (size_t)A_ * C_ * sizeof(bf16_t);    // 32 KB
    bf16_t* convbf = (bf16_t*)(ws + off); off += (size_t)B_ * T_ * C_ * sizeof(bf16_t); // 8 MB
    float*  scores = (float*)(ws + off);  off += (size_t)B_ * T_ * sizeof(float);     // 512 KB

    // 1) prep
    dec_proj_kernel<<<(B_ * A_ + 255) / 256, 256, 0, stream>>>(dec, W, bvec, decb);
    cvt_bf16_kernel<<<(A_ * H_ + 255) / 256, 256, 0, stream>>>(V, Vbf, A_ * H_);
    cvt_bf16_kernel<<<(A_ * C_ + 255) / 256, 256, 0, stream>>>(U, Ubf, A_ * C_);
    conv_feat_kernel<<<(B_ * T_ * C_ + 255) / 256, 256, 0, stream>>>(lalign, conv_w, conv_b, convbf);

    // 2) fused WMMA scores
    attn_scores_kernel<<<dim3(T_ / 64, B_), 256, 0, stream>>>(
        enc, Vbf, Ubf, convbf, decb, wvec, scores);

    // 3) softmax -> alignment (second output)
    softmax_kernel<<<B_, 256, 0, stream>>>(scores, align_out);

    // 4) context vector
    zero_out_kernel<<<(B_ * H_ + 511) / 512, 512, 0, stream>>>(out_vec);
    weighted_sum_kernel<<<dim3(B_, T_ / 512), 512, 0, stream>>>(enc, align_out, out_vec);
}
